// DistributedRingAttention_51531017617694
// MI455X (gfx1250) — matrix operand order repaired and checked by reading, not compiled, not run on hardware
//
#include <hip/hip_runtime.h>
#include <hip/hip_bf16.h>

typedef unsigned short u16;
typedef unsigned int   u32;
typedef __attribute__((ext_vector_type(16))) __bf16 v16bf;
typedef __attribute__((ext_vector_type(8)))  float  v8f;
typedef __attribute__((ext_vector_type(4)))  u32    v4u;
typedef __attribute__((ext_vector_type(2)))  u32    v2u;
typedef __attribute__((ext_vector_type(4)))  float  v4f;
typedef int v4i_vs __attribute__((vector_size(16)));   // matches builtin pointee

#define S_LEN 4096
#define HDIM  2048
#define NHEAD 16
#define HD    128
#define H3    6144

#if defined(__AMDGCN__) && __has_builtin(__builtin_amdgcn_global_load_async_to_lds_b128)
#define HAVE_ASYNC 1
#else
#define HAVE_ASYNC 0
#endif

__device__ __forceinline__ u16 f2bf(float f) {
    u32 u = __float_as_uint(f);
    u32 r = u + 0x7FFFu + ((u >> 16) & 1u);   // round-to-nearest-even
    return (u16)(r >> 16);
}

__device__ __forceinline__ v16bf load_frag2(const u16* p0, const u16* p1) {
    union { v16bf v; v4u q[2]; } f;
    f.q[0] = *(const v4u*)p0;
    f.q[1] = *(const v4u*)p1;
    return f.v;
}

__device__ __forceinline__ v8f wmma_bf16(v16bf a, v16bf b, v8f c) {
    return __builtin_amdgcn_wmma_f32_16x16x32_bf16(false, a, false, b,
                                                   (short)0, c, false, false);
}

__device__ __forceinline__ v8f zero8() {
    v8f z = {0.f, 0.f, 0.f, 0.f, 0.f, 0.f, 0.f, 0.f};
    return z;
}

// 16-byte global -> LDS copy; async (ASYNCcnt) when available.
__device__ __forceinline__ void cp16(const u16* g, u16* l) {
#if HAVE_ASYNC
    __builtin_amdgcn_global_load_async_to_lds_b128(
        (__attribute__((address_space(1))) v4i_vs*)g,
        (__attribute__((address_space(3))) v4i_vs*)l, 0, 0);
#else
    *(v4u*)l = *(const v4u*)g;
#endif
}

__device__ __forceinline__ void wait_async0() {
#if HAVE_ASYNC
#if __has_builtin(__builtin_amdgcn_s_wait_asynccnt)
    __builtin_amdgcn_s_wait_asynccnt(0);
#else
    __asm__ __volatile__("s_wait_asynccnt 0" ::: "memory");
#endif
#endif
}

// ---------------------------------------------------------------- convert ---
__global__ __launch_bounds__(256) void cvt_f32_bf16(const float* __restrict__ in,
                                                    u16* __restrict__ out, int n4) {
    int i = blockIdx.x * 256 + threadIdx.x;
    if (i >= n4) return;
    v4f f = ((const v4f*)in)[i];
    union { u16 s[4]; v2u q; } u;
    u.s[0] = f2bf(f[0]); u.s[1] = f2bf(f[1]);
    u.s[2] = f2bf(f[2]); u.s[3] = f2bf(f[3]);
    ((v2u*)out)[i] = u.q;
}

// --------------------------------------------------------- NT GEMM (bf16) ---
// C[M,N] = A[M,K] * B[N,K]^T ; A,B bf16 row-major (K contiguous), f32 acc.
// Block tile 256x128, 256 threads = 8 waves (4x2), wave tile 64x64.
// Double-buffered 32-wide K slabs with async global->LDS copies.
template <bool OUTF32>
__global__ __launch_bounds__(256)
void gemm_nt(const u16* __restrict__ A, const u16* __restrict__ Bm,
             float* __restrict__ Cf, u16* __restrict__ Cb,
             int M, int N, int K) {
    const int tidx = threadIdx.x;
    const int wave = tidx >> 5, lane = tidx & 31;
    const int hf = lane >> 4, l16 = lane & 15;
    const int bm = blockIdx.y * 256, bn = blockIdx.x * 128;
    const int wm = (wave & 3) * 64;   // wave row offset inside block tile
    const int wn = (wave >> 2) * 64;  // wave col offset

    __shared__ u16 As[2][256][40];
    __shared__ u16 Bs[2][128][40];

    v8f acc[4][4];
#pragma unroll
    for (int i = 0; i < 4; ++i)
#pragma unroll
        for (int j = 0; j < 4; ++j) acc[i][j] = zero8();

    const int r = tidx >> 2;            // 0..63
    const int c = (tidx & 3) << 3;      // 0,8,16,24

    auto issue = [&](int buf, int kk) {
        const u16* gA = A + (size_t)(bm + r) * K + kk + c;
#pragma unroll
        for (int p = 0; p < 4; ++p)
            cp16(gA + (size_t)(p * 64) * K, &As[buf][r + p * 64][c]);
        const u16* gB = Bm + (size_t)(bn + r) * K + kk + c;
#pragma unroll
        for (int p = 0; p < 2; ++p)
            cp16(gB + (size_t)(p * 64) * K, &Bs[buf][r + p * 64][c]);
    };

    const int nk = K >> 5;
    issue(0, 0);

    for (int t = 0; t < nk; ++t) {
        const int buf = t & 1;
        wait_async0();          // own async copies for `buf` complete
        __syncthreads();        // everyone's copies landed; prev reads done
        if (t + 1 < nk) issue(buf ^ 1, (t + 1) << 5);

        v16bf af[4], bf[4];
#pragma unroll
        for (int i = 0; i < 4; ++i) {
            const u16* arow = &As[buf][wm + i * 16 + l16][0];
            af[i] = load_frag2(arow + hf * 8, arow + 16 + hf * 8);
        }
#pragma unroll
        for (int j = 0; j < 4; ++j) {
            const u16* brow = &Bs[buf][wn + j * 16 + l16][0];
            bf[j] = load_frag2(brow + hf * 8, brow + 16 + hf * 8);
        }
#pragma unroll
        for (int i = 0; i < 4; ++i)
#pragma unroll
            for (int j = 0; j < 4; ++j)
                acc[i][j] = wmma_bf16(af[i], bf[j], acc[i][j]);
    }

#pragma unroll
    for (int i = 0; i < 4; ++i)
#pragma unroll
        for (int j = 0; j < 4; ++j)
#pragma unroll
            for (int v = 0; v < 8; ++v) {
                int row = bm + wm + i * 16 + v + 8 * hf;
                int col = bn + wn + j * 16 + l16;
                float val = acc[i][j][v];
                if constexpr (OUTF32) Cf[(size_t)row * N + col] = val;
                else                  Cb[(size_t)row * N + col] = f2bf(val);
            }
}

// ----------------------------------------------------------- attention -------
// One WG (8 waves) = head blockIdx.y, queries [blockIdx.x*128, +128).
// Each wave owns 16 query rows; flash loop over 64-key blocks.
__global__ __launch_bounds__(256)
void attn_kernel(const u16* __restrict__ qkv, u16* __restrict__ outb) {
    const int tidx = threadIdx.x;
    const int wave = tidx >> 5, lane = tidx & 31;
    const int hf = lane >> 4, l16 = lane & 15;
    const int head = blockIdx.y;
    const int qb = blockIdx.x * 128;
    const float scale = 0.0883883476483184f;  // 1/sqrt(128)

    __shared__ u16 Ks[64][136];      // [key][d] row-major
    __shared__ u16 Vt[128][72];      // [d][key] transposed
    __shared__ u16 Ps[8][16][72];    // per-wave P staging (C-layout -> A-layout)

    // Q fragments straight from global (A-frag chunks contiguous in Q row)
    v16bf qf[4];
    {
        const u16* qrow = qkv + (size_t)(qb + wave * 16 + l16) * H3 + head * HD;
#pragma unroll
        for (int ks = 0; ks < 4; ++ks)
            qf[ks] = load_frag2(qrow + ks * 32 + hf * 8,
                                qrow + ks * 32 + 16 + hf * 8);
    }

    float m_i[8], l_i[8];
    v8f O[8];
#pragma unroll
    for (int v = 0; v < 8; ++v) { m_i[v] = -1e30f; l_i[v] = 0.f; }
#pragma unroll
    for (int jo = 0; jo < 8; ++jo) O[jo] = zero8();

    for (int kb = 0; kb < S_LEN; kb += 64) {
        __syncthreads();
        // K block async (row-major); V block transposed via DS stores
#pragma unroll
        for (int cc = 0; cc < 4; ++cc) {
            int idx = cc * 256 + tidx;            // 0..1023
            int row = idx >> 4;                   // 0..63 key
            int col = (idx & 15) << 3;            // 0..120 d
            const u16* srcK = qkv + (size_t)(kb + row) * H3 + HDIM + head * HD + col;
            cp16(srcK, &Ks[row][col]);
            const u16* srcV = qkv + (size_t)(kb + row) * H3 + 2 * HDIM + head * HD + col;
            union { v4u q; u16 s[8]; } u;
            u.q = *(const v4u*)srcV;
#pragma unroll
            for (int i = 0; i < 8; ++i) Vt[col + i][row] = u.s[i];
        }
        wait_async0();
        __syncthreads();

        // scores = Q * K^T  (4 n-tiles of 16 keys, 4 k-steps of 32 dims)
        v8f sc[4];
#pragma unroll
        for (int j = 0; j < 4; ++j) {
            v8f a = zero8();
#pragma unroll
            for (int ks = 0; ks < 4; ++ks) {
                const u16* brow = &Ks[j * 16 + l16][ks * 32 + hf * 8];
                a = wmma_bf16(qf[ks], load_frag2(brow, brow + 16), a);
            }
            sc[j] = a;
        }

        // online softmax (row scalars per v, replicated across 16-lane group)
#pragma unroll
        for (int v = 0; v < 8; ++v) {
            float mx = -1e30f;
#pragma unroll
            for (int j = 0; j < 4; ++j) { sc[j][v] *= scale; mx = fmaxf(mx, sc[j][v]); }
            mx = fmaxf(mx, __shfl_xor(mx, 1, 32));
            mx = fmaxf(mx, __shfl_xor(mx, 2, 32));
            mx = fmaxf(mx, __shfl_xor(mx, 4, 32));
            mx = fmaxf(mx, __shfl_xor(mx, 8, 32));
            float mn = fmaxf(m_i[v], mx);
            float alpha = __expf(m_i[v] - mn);
            float rs = 0.f;
#pragma unroll
            for (int j = 0; j < 4; ++j) {
                float e = __expf(sc[j][v] - mn);
                sc[j][v] = e; rs += e;
            }
            rs += __shfl_xor(rs, 1, 32);
            rs += __shfl_xor(rs, 2, 32);
            rs += __shfl_xor(rs, 4, 32);
            rs += __shfl_xor(rs, 8, 32);
            l_i[v] = l_i[v] * alpha + rs;
            m_i[v] = mn;
#pragma unroll
            for (int jo = 0; jo < 8; ++jo) O[jo][v] *= alpha;
        }

        // stage P (C-layout -> LDS row-major), then read back as A-fragments
#pragma unroll
        for (int j = 0; j < 4; ++j)
#pragma unroll
            for (int v = 0; v < 8; ++v)
                Ps[wave][v + 8 * hf][j * 16 + l16] = f2bf(sc[j][v]);
        // wave-private buffer: DS is in-order per wave; force order vs loads
        __asm__ __volatile__("s_wait_dscnt 0" ::: "memory");

        v16bf pa[2];
#pragma unroll
        for (int k2 = 0; k2 < 2; ++k2) {
            const u16* prow = &Ps[wave][l16][0];
            pa[k2] = load_frag2(prow + k2 * 32 + hf * 8,
                                prow + k2 * 32 + 16 + hf * 8);
        }
        // O += P * V  (8 n-tiles of dims, 2 k-steps of 32 keys)
#pragma unroll
        for (int jo = 0; jo < 8; ++jo)
#pragma unroll
            for (int k2 = 0; k2 < 2; ++k2) {
                const u16* brow = &Vt[jo * 16 + l16][k2 * 32 + hf * 8];
                O[jo] = wmma_bf16(pa[k2], load_frag2(brow, brow + 16), O[jo]);
            }
    }

#pragma unroll
    for (int v = 0; v < 8; ++v) l_i[v] = 1.0f / (l_i[v] + 1e-8f);
#pragma unroll
    for (int jo = 0; jo < 8; ++jo)
#pragma unroll
        for (int v = 0; v < 8; ++v) {
            int row = qb + wave * 16 + v + 8 * hf;
            int col = head * HD + jo * 16 + l16;
            outb[(size_t)row * HDIM + col] = f2bf(O[jo][v] * l_i[v]);
        }
}

// ---------------------------------------------------------------- launch -----
extern "C" void kernel_launch(void* const* d_in, const int* in_sizes, int n_in,
                              void* d_out, int out_size, void* d_ws, size_t ws_size,
                              hipStream_t stream) {
    (void)in_sizes; (void)n_in; (void)out_size; (void)ws_size;
    const float* x    = (const float*)d_in[0];
    const float* wqkv = (const float*)d_in[1];
    const float* wout = (const float*)d_in[2];
    float* out = (float*)d_out;
    char* ws = (char*)d_ws;

    u16* xb    = (u16*)(ws);                            // 4096x2048 bf16 (16 MiB)
    u16* wqkvb = (u16*)(ws + ((size_t)16 << 20));       // 6144x2048 bf16 (24 MiB)
    u16* woutb = (u16*)(ws + ((size_t)40 << 20));       // 2048x2048 bf16 ( 8 MiB)
    u16* qkvb  = (u16*)(ws + ((size_t)48 << 20));       // 4096x6144 bf16 (48 MiB)
    u16* attb  = xb;                                    // reuse x region after GEMM1

    {
        int n4 = (S_LEN * HDIM) / 4;
        cvt_f32_bf16<<<(n4 + 255) / 256, 256, 0, stream>>>(x, xb, n4);
    }
    {
        int n4 = (H3 * HDIM) / 4;
        cvt_f32_bf16<<<(n4 + 255) / 256, 256, 0, stream>>>(wqkv, wqkvb, n4);
    }
    {
        int n4 = (HDIM * HDIM) / 4;
        cvt_f32_bf16<<<(n4 + 255) / 256, 256, 0, stream>>>(wout, woutb, n4);
    }

    // qkv = x * w_qkv^T
    gemm_nt<false><<<dim3(H3 / 128, S_LEN / 256), 256, 0, stream>>>(
        xb, wqkvb, nullptr, qkvb, S_LEN, H3, HDIM);

    // flash attention
    attn_kernel<<<dim3(S_LEN / 128, NHEAD), 256, 0, stream>>>(qkvb, attb);

    // out = att * w_out^T  (f32 output)
    gemm_nt<true><<<dim3(HDIM / 128, S_LEN / 256), 256, 0, stream>>>(
        attb, woutb, out, nullptr, S_LEN, HDIM, HDIM);
}
